// FlashFFN_78786880078264
// MI455X (gfx1250) — matrix operand order repaired and checked by reading, not compiled, not run on hardware
//
#include <hip/hip_runtime.h>
#include <cstddef>

// ---------------------------------------------------------------------------
// Fused SwiGLU FFN for MI455X (gfx1250, wave32, WMMA bf16 16x16x32)
//   y = (SiLU(x @ Wg^T) * (x @ Wu^T)) @ Wd^T
// x:[M=8192,H=1024] f32, Wg/Wu:[I=4096,H] f32, Wd:[H,I] f32, y:[M,H] f32
//
// Pass 0 (if d_ws is big enough): convert Wg/Wu/Wd to bf16 in d_ws once
//   (24 MiB; ~3us at HBM rate). bf16 weights fit in 192MB L2 -> all 256
//   workgroups stream them from L2 with no per-WG cvt VALU cost.
// Pass 1: fused FFN. x tile staged once in LDS as bf16; h_mid never touches
//   HBM; y accumulators live in VGPRs across the whole I loop.
// ---------------------------------------------------------------------------

typedef __attribute__((ext_vector_type(16))) __bf16 v16bf;
typedef __attribute__((ext_vector_type(8)))  __bf16 v8bf;
typedef __attribute__((ext_vector_type(4)))  __bf16 v4bf;
typedef __attribute__((ext_vector_type(8)))  float  v8f;

#define FFN_H      1024
#define FFN_I      4096
#define M_TILE     32
#define I_TILE     64
#define THREADS    512        // 16 waves of 32

// padded LDS strides (conflict-free b128 lane patterns)
#define XS   (FFN_H + 8)      // bf16 elems per sX row   (2064 B, 16B aligned)
#define GUS  (I_TILE + 4)     // f32  elems per sGU row
#define MS   (I_TILE + 8)     // bf16 elems per sMid row (144 B, 16B aligned)

#define SX_BYTES   (M_TILE * XS * 2)
#define SGU_BYTES  (2 * M_TILE * GUS * 4)
#define SMID_BYTES (M_TILE * MS * 2)
#define SMEM_BYTES (SX_BYTES + SGU_BYTES + SMID_BYTES)

#define WMMA_BF16(a, b, c) \
    __builtin_amdgcn_wmma_f32_16x16x32_bf16(false, (a), false, (b), (short)0, (c), false, false)

// A-fragment (16x32 bf16) from LDS, row-major source with stride `stride`.
// Lanes 0-15: row=l16, K in {kb..kb+15}; lanes 16-31: K +16.
// (kcol already includes hi*16.)
static __device__ __forceinline__ v16bf load_a_lds(const __bf16* base, int stride,
                                                   int row, int kcol) {
    const v8bf c0 = *(const v8bf*)(base + row * stride + kcol);
    const v8bf c1 = *(const v8bf*)(base + row * stride + kcol + 8);
    v16bf a;
#pragma unroll
    for (int i = 0; i < 8; ++i) { a[i] = c0[i]; a[8 + i] = c1[i]; }
    return a;
}

// B-fragment (32x16): lane holds column n = lane&15 (one weight row),
// 16 contiguous K values (caller adds hi*16 to the K base).
// f32 source path: 4 x b128 loads + cvt.
static __device__ __forceinline__ v16bf load_b(const float* __restrict__ p) {
    v16bf b;
#pragma unroll
    for (int q = 0; q < 4; ++q) {
        const float4 v = ((const float4*)p)[q];
        b[q * 4 + 0] = (__bf16)v.x;
        b[q * 4 + 1] = (__bf16)v.y;
        b[q * 4 + 2] = (__bf16)v.z;
        b[q * 4 + 3] = (__bf16)v.w;
    }
    return b;
}
// bf16 source path: 2 x b128 loads, zero cvt.
static __device__ __forceinline__ v16bf load_b(const __bf16* __restrict__ p) {
    const v8bf c0 = *(const v8bf*)p;
    const v8bf c1 = *(const v8bf*)(p + 8);
    v16bf b;
#pragma unroll
    for (int i = 0; i < 8; ++i) { b[i] = c0[i]; b[8 + i] = c1[i]; }
    return b;
}

// ---- Pass 0: f32 -> bf16 weight conversion (8 floats / thread) ----------
__global__ __launch_bounds__(256)
void convert_w_kernel(const float* __restrict__ src, __bf16* __restrict__ dst,
                      int n8) {
    const int i = blockIdx.x * blockDim.x + threadIdx.x;
    if (i >= n8) return;
    const float4 a = ((const float4*)src)[2 * i + 0];
    const float4 b = ((const float4*)src)[2 * i + 1];
    v8bf o;
    o[0] = (__bf16)a.x; o[1] = (__bf16)a.y; o[2] = (__bf16)a.z; o[3] = (__bf16)a.w;
    o[4] = (__bf16)b.x; o[5] = (__bf16)b.y; o[6] = (__bf16)b.z; o[7] = (__bf16)b.w;
    ((v8bf*)dst)[i] = o;
}

// ---- Pass 1: fused FFN (WT = __bf16 normally, float fallback) -----------
template <typename WT>
__global__ __launch_bounds__(THREADS, 1)
void flash_ffn_swiglu_kernel(const float* __restrict__ x,
                             const WT* __restrict__ wg,
                             const WT* __restrict__ wu,
                             const WT* __restrict__ wd,
                             float* __restrict__ y) {
    extern __shared__ char smem[];
    __bf16* sX   = (__bf16*)smem;                          // [M_TILE][XS]
    float*  sGU  = (float*)(smem + SX_BYTES);              // [2][M_TILE][GUS]
    __bf16* sMid = (__bf16*)(smem + SX_BYTES + SGU_BYTES); // [M_TILE][MS]

    const int tid  = threadIdx.x;
    const int wave = tid >> 5;
    const int lane = tid & 31;
    const int l16  = lane & 15;
    const int hi   = lane >> 4;              // 0: lanes 0-15, 1: lanes 16-31
    const int m0   = blockIdx.x * M_TILE;

    // ---- Stage x tile into LDS as bf16 (read once from HBM) -------------
    for (int idx = tid; idx < M_TILE * (FFN_H / 4); idx += THREADS) {
        const int row = idx >> 8;            // H/4 = 256 float4 per row
        const int c4  = idx & 255;
        const float4 v = ((const float4*)(x + (size_t)(m0 + row) * FFN_H))[c4];
        v4bf o;
        o[0] = (__bf16)v.x; o[1] = (__bf16)v.y;
        o[2] = (__bf16)v.z; o[3] = (__bf16)v.w;
        *(v4bf*)(&sX[row * XS + c4 * 4]) = o;
    }
    __syncthreads();

    // GEMM1 role: matrix (gate/up) x n-subtile x m-subtile, one tile per wave
    const int g1_mat  = wave & 1;
    const int g1_nsub = (wave >> 1) & 3;
    const int g1_msub = (wave >> 3) & 1;
    const WT* __restrict__ wmat = g1_mat ? wu : wg;

    // GEMM2: wave owns 4 n-subtiles x 2 m-subtiles; y stays in VGPRs.
    v8f yAcc[4][2];
#pragma unroll
    for (int n = 0; n < 4; ++n)
#pragma unroll
        for (int m = 0; m < 2; ++m) yAcc[n][m] = {};

    for (int ic = 0; ic < FFN_I / I_TILE; ++ic) {
        const int ibase = ic * I_TILE;

        // ---- GEMM1: one 16x16 (gate|up) tile per wave, K = H ------------
        v8f accGU = {};
        const WT* __restrict__ wrow =
            wmat + (size_t)(ibase + g1_nsub * 16 + l16) * FFN_H;
#pragma unroll 4
        for (int ks = 0; ks < FFN_H / 32; ++ks) {
            const int k = ks * 32;
            const v16bf a = load_a_lds(sX, XS, g1_msub * 16 + l16, k + hi * 16);
            const v16bf b = load_b(wrow + k + hi * 16);
            accGU = WMMA_BF16(a, b, accGU);
        }
        // Park gate/up tile in LDS (C/D layout: VGPR r -> M = r + hi*8, N = l16)
        {
            float* gu = sGU + g1_mat * (M_TILE * GUS);
#pragma unroll
            for (int r = 0; r < 8; ++r)
                gu[(g1_msub * 16 + r + hi * 8) * GUS + g1_nsub * 16 + l16] = accGU[r];
        }
        __syncthreads();

        // ---- Elementwise SiLU(g) * u -> bf16 sMid -----------------------
#pragma unroll
        for (int e = 0; e < (M_TILE * I_TILE) / THREADS; ++e) {
            const int idx = e * THREADS + tid;
            const int row = idx >> 6;        // / I_TILE
            const int col = idx & (I_TILE - 1);
            const float g = sGU[row * GUS + col];
            const float u = sGU[M_TILE * GUS + row * GUS + col];
            // silu(g) = g * sigmoid(g); fast rcp (v_rcp_f32), no IEEE div chain
            const float s = g * __builtin_amdgcn_rcpf(1.0f + __expf(-g));
            sMid[row * MS + col] = (__bf16)(s * u);
        }
        __syncthreads();

        // ---- GEMM2: y[32 x 1024] += sMid[32 x 64] @ Wd[:, chunk]^T ------
        v16bf a2[2][2];
#pragma unroll
        for (int m = 0; m < 2; ++m)
#pragma unroll
            for (int ks = 0; ks < 2; ++ks)
                a2[m][ks] = load_a_lds(sMid, MS, m * 16 + l16, ks * 32 + hi * 16);

#pragma unroll
        for (int n = 0; n < 4; ++n) {
            const int hrow = (wave * 4 + n) * 16 + l16;
            const WT* __restrict__ wdr = wd + (size_t)hrow * FFN_I + ibase;
            const v16bf b0 = load_b(wdr + hi * 16);
            const v16bf b1 = load_b(wdr + 32 + hi * 16);
#pragma unroll
            for (int m = 0; m < 2; ++m) {
                yAcc[n][m] = WMMA_BF16(a2[m][0], b0, yAcc[n][m]);
                yAcc[n][m] = WMMA_BF16(a2[m][1], b1, yAcc[n][m]);
            }
        }
        // No barrier needed here: next-iteration LDS writes are fenced by
        // the two barriers above (all waves must pass GEMM2 first).
    }

    // ---- Write y tiles (f32) -------------------------------------------
#pragma unroll
    for (int n = 0; n < 4; ++n) {
        const int hcol = (wave * 4 + n) * 16 + l16;
#pragma unroll
        for (int m = 0; m < 2; ++m) {
#pragma unroll
            for (int r = 0; r < 8; ++r) {
                const int row = m0 + m * 16 + r + hi * 8;
                y[(size_t)row * FFN_H + hcol] = yAcc[n][m][r];
            }
        }
    }
}

extern "C" void kernel_launch(void* const* d_in, const int* in_sizes, int n_in,
                              void* d_out, int out_size, void* d_ws, size_t ws_size,
                              hipStream_t stream) {
    (void)n_in; (void)out_size;
    const float* x  = (const float*)d_in[0];
    const float* wg = (const float*)d_in[1];
    const float* wu = (const float*)d_in[2];
    const float* wd = (const float*)d_in[3];
    float* y = (float*)d_out;

    const int M    = in_sizes[0] / FFN_H;    // B*S = 8192
    const int grid = M / M_TILE;             // 256 workgroups

    const size_t WELEMS  = (size_t)FFN_I * FFN_H;        // 4Mi per matrix
    const size_t WS_NEED = 3 * WELEMS * sizeof(__bf16);  // 24 MiB

    if (ws_size >= WS_NEED) {
        // Pass 0: one-shot weight downconvert into workspace.
        __bf16* wgb = (__bf16*)d_ws;
        __bf16* wub = wgb + WELEMS;
        __bf16* wdb = wub + WELEMS;
        const int n8 = (int)(WELEMS / 8);                // 524288 threads
        const int cg = n8 / 256;
        convert_w_kernel<<<cg, 256, 0, stream>>>(wg, wgb, n8);
        convert_w_kernel<<<cg, 256, 0, stream>>>(wu, wub, n8);
        convert_w_kernel<<<cg, 256, 0, stream>>>(wd, wdb, n8);

        hipFuncSetAttribute(
            reinterpret_cast<const void*>(&flash_ffn_swiglu_kernel<__bf16>),
            hipFuncAttributeMaxDynamicSharedMemorySize, SMEM_BYTES);
        flash_ffn_swiglu_kernel<__bf16>
            <<<grid, THREADS, SMEM_BYTES, stream>>>(x, wgb, wub, wdb, y);
    } else {
        // Fallback: convert weights on the fly (no workspace requirement).
        hipFuncSetAttribute(
            reinterpret_cast<const void*>(&flash_ffn_swiglu_kernel<float>),
            hipFuncAttributeMaxDynamicSharedMemorySize, SMEM_BYTES);
        flash_ffn_swiglu_kernel<float>
            <<<grid, THREADS, SMEM_BYTES, stream>>>(x, wg, wu, wd, y);
    }
}
